// MultiHeadAttention_80272938762958
// MI455X (gfx1250) — matrix operand order repaired and checked by reading, not compiled, not run on hardware
//
#include <hip/hip_runtime.h>

// ---------------------------------------------------------------------------
// MI455X (gfx1250) multi-head attention, wave32 + v_wmma_f32_16x16x32_f16.
// Bandwidth-bound: the [H,B,L,L] f32 attention output (256MB) dominates, so
// all GEMMs run in f16/f32-acc WMMA. Weights/operands are pre-transposed so
// every WMMA B-fragment is one contiguous 32B load per lane and every
// A-fragment is two contiguous 16B loads per lane. GEMM waves compute 16x64
// blocks (4 accumulators sharing each A-fragment) and are software-pipelined:
// next iteration's fragments are loaded before this iteration's WMMAs so
// s_wait_loadcnt only covers loads issued a full iteration earlier.
// ---------------------------------------------------------------------------

typedef __attribute__((ext_vector_type(16))) _Float16 v16h;
typedef __attribute__((ext_vector_type(8)))  _Float16 v8h;
typedef __attribute__((ext_vector_type(4)))  _Float16 v4h;
typedef __attribute__((ext_vector_type(8)))  float    v8f;

#define D_MODEL 1024
#define NB      4
#define LSEQ    1024
#define NH      16
#define DH      64

__device__ __forceinline__ v8f wmma16(v16h a, v16h b, v8f c) {
    return __builtin_amdgcn_wmma_f32_16x16x32_f16(
        /*neg_a=*/false, a, /*neg_b=*/false, b,
        /*c_mod=*/(short)0, c, /*reuse_a=*/false, /*reuse_b=*/false);
}

// A-fragment: row-major A, `arow` points at this lane's row (lane%16 of tile).
// Halves 0..15 = K[k0+kb .. +15], kb = (lane>>4)*16.
__device__ __forceinline__ v16h load_a_row(const _Float16* __restrict__ arow,
                                           int k0, int lane) {
    const _Float16* p = arow + k0 + ((lane >> 4) << 4);
    v8h lo = *(const v8h*)p;
    v8h hi = *(const v8h*)(p + 8);
    return __builtin_shufflevector(lo, hi, 0,1,2,3,4,5,6,7,8,9,10,11,12,13,14,15);
}

// B-fragment from transposed-B (row-major [N][K]); `brow` = lane's column row.
// Halves 0..15 = K[k0 + (lane>>4)*16 .. +15] — one contiguous 32B load.
__device__ __forceinline__ v16h load_bt_row(const _Float16* __restrict__ brow,
                                            int k0, int lane) {
    return *(const v16h*)(brow + k0 + ((lane >> 4) << 4));
}

// Software-pipelined 16x64 x K=1024 GEMM: 4 accumulators, double-buffered
// fragments so loads for step k+1 are in flight during step k's WMMAs.
__device__ __forceinline__ void gemm_16x64_k1024(
    const _Float16* __restrict__ arow,
    const _Float16* __restrict__ b0r, const _Float16* __restrict__ b1r,
    const _Float16* __restrict__ b2r, const _Float16* __restrict__ b3r,
    int lane, v8f acc[4]) {
    v16h a  = load_a_row(arow, 0, lane);
    v16h b0 = load_bt_row(b0r, 0, lane);
    v16h b1 = load_bt_row(b1r, 0, lane);
    v16h b2 = load_bt_row(b2r, 0, lane);
    v16h b3 = load_bt_row(b3r, 0, lane);
    v8f c0 = {}, c1 = {}, c2 = {}, c3 = {};
    #pragma unroll 1
    for (int k0 = 32; k0 < D_MODEL; k0 += 32) {
        v16h an  = load_a_row(arow, k0, lane);      // prefetch next step
        v16h bn0 = load_bt_row(b0r, k0, lane);
        v16h bn1 = load_bt_row(b1r, k0, lane);
        v16h bn2 = load_bt_row(b2r, k0, lane);
        v16h bn3 = load_bt_row(b3r, k0, lane);
        c0 = wmma16(a, b0, c0);                     // consume current step
        c1 = wmma16(a, b1, c1);
        c2 = wmma16(a, b2, c2);
        c3 = wmma16(a, b3, c3);
        a = an; b0 = bn0; b1 = bn1; b2 = bn2; b3 = bn3;
    }
    c0 = wmma16(a, b0, c0);                         // peeled last step
    c1 = wmma16(a, b1, c1);
    c2 = wmma16(a, b2, c2);
    c3 = wmma16(a, b3, c3);
    acc[0] = c0; acc[1] = c1; acc[2] = c2; acc[3] = c3;
}

// ------------------------- conversion kernels ------------------------------

__global__ __launch_bounds__(256) void cvt_f32_to_f16(const float* __restrict__ src,
                                                      _Float16* __restrict__ dst,
                                                      int n4) {
    int i = blockIdx.x * blockDim.x + threadIdx.x;
    if (i < n4) {
        float4 f = ((const float4*)src)[i];
        v4h h = { (_Float16)f.x, (_Float16)f.y, (_Float16)f.z, (_Float16)f.w };
        ((v4h*)dst)[i] = h;
    }
}

// out[c*R + r] = (f16) in[r*C + c]   (batched over blockIdx.y)
__global__ __launch_bounds__(256) void transpose_cvt(const float* __restrict__ in,
                                                     _Float16* __restrict__ out,
                                                     int R, int C) {
    size_t base = (size_t)blockIdx.y * R * C;
    int idx = blockIdx.x * blockDim.x + threadIdx.x;
    if (idx < R * C) {
        int c = idx / R;
        int r = idx - c * R;
        out[base + idx] = (_Float16)in[base + (size_t)r * C + c];
    }
}

// ------------------------- projection GEMM ---------------------------------
// X16 [4096][1024] @ WT [1024(n=h*64+e)][1024(d)] -> per-head q/k/v (+bias).
// One wave computes a 16x64 block == one full head slice of 16 rows.
// mat 0: q16[b][h][l][e]  mat 1: k16[b][h][l][e]  mat 2: vT16[b][h][e][l]
__global__ __launch_bounds__(256) void proj_kernel(
    const _Float16* __restrict__ X0, const _Float16* __restrict__ X1,
    const _Float16* __restrict__ X2,
    const _Float16* __restrict__ WT0, const _Float16* __restrict__ WT1,
    const _Float16* __restrict__ WT2,
    const float* __restrict__ bq, const float* __restrict__ bk,
    const float* __restrict__ bv,
    _Float16* __restrict__ q16, _Float16* __restrict__ k16,
    _Float16* __restrict__ vT16) {
    int lane = threadIdx.x & 31;
    int wid  = blockIdx.x * 8 + (threadIdx.x >> 5);   // 0 .. 3*4096-1
    int mat  = wid >> 12;                              // 4096 jobs per matrix
    int t    = wid & 4095;
    int m0   = (t >> 4) << 4;                          // 256 m-tiles
    int h    = t & 15;                                 // n-group == head
    int n0   = h << 6;                                 // 64-wide block

    const _Float16* X  = (mat == 0) ? X0  : (mat == 1) ? X1  : X2;
    const _Float16* WT = (mat == 0) ? WT0 : (mat == 1) ? WT1 : WT2;
    const float*  bias = (mat == 0) ? bq  : (mat == 1) ? bk  : bv;

    const _Float16* arow = X + (size_t)(m0 + (lane & 15)) * D_MODEL;
    const _Float16* bbase = WT + (size_t)(n0 + (lane & 15)) * D_MODEL;

    v8f acc[4];
    gemm_16x64_k1024(arow, bbase, bbase + 16 * D_MODEL, bbase + 32 * D_MODEL,
                     bbase + 48 * D_MODEL, lane, acc);

    int bidx = m0 >> 10;            // batch
    int l0   = m0 & 1023;
    int mlo  = (lane >> 4) << 3;

    if (mat < 2) {
        _Float16* dst = (mat == 0) ? q16 : k16;
        size_t base = ((size_t)bidx * NH + h) * LSEQ;
        #pragma unroll
        for (int s = 0; s < 4; ++s) {
            int e = (s << 4) + (lane & 15);
            float bn = bias[n0 + e];
            #pragma unroll
            for (int v = 0; v < 8; ++v)
                dst[(base + l0 + mlo + v) * DH + e] = (_Float16)(acc[s][v] + bn);
        }
    } else {
        #pragma unroll
        for (int s = 0; s < 4; ++s) {
            int e = (s << 4) + (lane & 15);
            float bn = bias[n0 + e];
            size_t base = (((size_t)bidx * NH + h) * DH + e) * LSEQ + l0 + mlo;
            v8h outp;
            #pragma unroll
            for (int v = 0; v < 8; ++v) outp[v] = (_Float16)(acc[s][v] + bn);
            *(v8h*)(vT16 + base) = outp;   // contiguous 16B store
        }
    }
}

// ---------------- fused scores + softmax + P@V per (b,h,qtile) -------------
__global__ __launch_bounds__(256) void attn_kernel(
    const _Float16* __restrict__ q16, const _Float16* __restrict__ k16,
    const _Float16* __restrict__ vT16,
    float* __restrict__ attn_out,           // [H][B][L][L]
    _Float16* __restrict__ O16) {           // [B][L][H*DH]
    __shared__ _Float16 S16[16][LSEQ];      // scores, then probs in-place (32KB)
    __shared__ float    red[16][16];
    __shared__ float    part[16][DH];       // phase-3 K-split partials (4KB)

    int lane = threadIdx.x & 31;
    int wave = threadIdx.x >> 5;            // 0..7
    int blk  = blockIdx.x;                  // ((b*16+h)*64 + qtile)
    int qt   = blk & 63;
    int bh   = blk >> 6;
    int b    = bh >> 4;
    int h    = bh & 15;
    int q0   = qt << 4;

    const _Float16* qb  = q16  + ((size_t)bh * LSEQ) * DH;
    const _Float16* kb  = k16  + ((size_t)bh * LSEQ) * DH;
    const _Float16* vtb = vT16 + ((size_t)bh * DH) * LSEQ;

    // Phase 1: S = (q . k^T) / 8 for 16 queries x 1024 keys; 8 key-tiles/wave.
    // B fragments for tile i+1 are prefetched while tile i's WMMAs run.
    const _Float16* arow = qb + (size_t)(q0 + (lane & 15)) * DH;
    v16h a0 = load_a_row(arow, 0, lane);
    v16h a1 = load_a_row(arow, 32, lane);
    int mlo = (lane >> 4) << 3;
    {
        const _Float16* brp = kb + (size_t)(((wave * 8) << 4) + (lane & 15)) * DH;
        v16h bb0 = load_bt_row(brp, 0, lane);
        v16h bb1 = load_bt_row(brp, 32, lane);
        #pragma unroll
        for (int i = 0; i < 8; ++i) {
            v16h nb0, nb1;
            if (i < 7) {
                const _Float16* nx = brp + (size_t)(i + 1) * 16 * DH;
                nb0 = load_bt_row(nx, 0, lane);
                nb1 = load_bt_row(nx, 32, lane);
            }
            v8f acc = {};
            acc = wmma16(a0, bb0, acc);
            acc = wmma16(a1, bb1, acc);
            int col = ((wave * 8 + i) << 4) + (lane & 15);
            #pragma unroll
            for (int v = 0; v < 8; ++v)
                S16[mlo + v][col] = (_Float16)(acc[v] * 0.125f);
            if (i < 7) { bb0 = nb0; bb1 = nb1; }
        }
    }
    __syncthreads();

    // Phase 2: softmax, 256 threads: row = tid>>4, 64-wide segment = tid&15
    int row = threadIdx.x >> 4;
    int seg = threadIdx.x & 15;
    _Float16* srow = &S16[row][seg * 64];

    float lmax = -1e30f;
    #pragma unroll 8
    for (int j = 0; j < 64; ++j) lmax = fmaxf(lmax, (float)srow[j]);
    red[row][seg] = lmax;
    __syncthreads();
    float rmax = red[row][0];
    #pragma unroll
    for (int j = 1; j < 16; ++j) rmax = fmaxf(rmax, red[row][j]);

    float lsum = 0.f;
    #pragma unroll 8
    for (int j = 0; j < 64; ++j) lsum += __expf((float)srow[j] - rmax);
    __syncthreads();
    red[row][seg] = lsum;
    __syncthreads();
    float rsum = 0.f;
    #pragma unroll
    for (int j = 0; j < 16; ++j) rsum += red[row][j];
    float rinv = 1.0f / rsum;

    size_t abase = ((((size_t)h * NB + b) * LSEQ) + (q0 + row)) * LSEQ + seg * 64;
    #pragma unroll 4
    for (int j = 0; j < 64; ++j) {
        float p = __expf((float)srow[j] - rmax) * rinv;
        attn_out[abase + j] = p;        // f32 attention output (dominant traffic)
        srow[j] = (_Float16)p;          // in-place probs for P@V (same thread)
    }
    __syncthreads();

    // Phase 3: out_h = P @ v  (16x64, K=1024). 8 waves: ntile = wave&3,
    // K-half = wave>>2 (512 keys each); upper halves combined through LDS.
    // LDS A-fragment + global B-fragment for step k+1 prefetched during step k.
    {
        int n     = ((wave & 3) << 4) + (lane & 15);     // e index 0..63
        int kbase = (wave >> 2) << 9;                    // 0 or 512
        const _Float16* brow = vtb + (size_t)n * LSEQ;   // v^T row-major == B^T
        const _Float16* prow = &S16[lane & 15][0];
        auto load_p = [&](int k0) -> v16h {
            const _Float16* p = prow + k0 + ((lane >> 4) << 4);
            v8h lo = *(const v8h*)p;                     // ds_load_b128
            v8h hi = *(const v8h*)(p + 8);
            return __builtin_shufflevector(lo, hi,
                   0,1,2,3,4,5,6,7,8,9,10,11,12,13,14,15);
        };
        v16h pa = load_p(kbase);
        v16h pb = load_bt_row(brow, kbase, lane);
        v8f acc = {};
        #pragma unroll 1
        for (int kk = 32; kk < 512; kk += 32) {
            v16h na = load_p(kbase + kk);
            v16h nb = load_bt_row(brow, kbase + kk, lane);
            acc = wmma16(pa, pb, acc);
            pa = na; pb = nb;
        }
        acc = wmma16(pa, pb, acc);

        if (wave >= 4) {
            #pragma unroll
            for (int v = 0; v < 8; ++v) part[mlo + v][n] = acc[v];
        }
        __syncthreads();
        if (wave < 4) {
            #pragma unroll
            for (int v = 0; v < 8; ++v) {
                float r = acc[v] + part[mlo + v][n];
                size_t off = ((size_t)b * LSEQ + q0 + mlo + v) * D_MODEL
                           + h * DH + n;
                O16[off] = (_Float16)r;
            }
        }
    }
}

// ------------------------- output projection -------------------------------
// One wave computes a 16x64 block of out = O16 @ Wo + bo.
__global__ __launch_bounds__(256) void outproj_kernel(
    const _Float16* __restrict__ O16, const _Float16* __restrict__ WoT,
    const float* __restrict__ bo, float* __restrict__ out) {
    int lane = threadIdx.x & 31;
    int wid  = blockIdx.x * 8 + (threadIdx.x >> 5);  // 0..4095
    int m0   = (wid >> 4) << 4;
    int n0   = (wid & 15) << 6;

    const _Float16* arow  = O16 + (size_t)(m0 + (lane & 15)) * D_MODEL;
    const _Float16* bbase = WoT + (size_t)(n0 + (lane & 15)) * D_MODEL;

    v8f acc[4];
    gemm_16x64_k1024(arow, bbase, bbase + 16 * D_MODEL, bbase + 32 * D_MODEL,
                     bbase + 48 * D_MODEL, lane, acc);

    int mlo = (lane >> 4) << 3;
    #pragma unroll
    for (int s = 0; s < 4; ++s) {
        int n = n0 + (s << 4) + (lane & 15);
        float bn = bo[n];
        #pragma unroll
        for (int v = 0; v < 8; ++v)
            out[(size_t)(m0 + mlo + v) * D_MODEL + n] = acc[s][v] + bn;
    }
}

// ---------------------------------------------------------------------------

extern "C" void kernel_launch(void* const* d_in, const int* in_sizes, int n_in,
                              void* d_out, int out_size, void* d_ws, size_t ws_size,
                              hipStream_t stream) {
    const float* Q  = (const float*)d_in[0];
    const float* K  = (const float*)d_in[1];
    const float* V  = (const float*)d_in[2];
    const float* Wq = (const float*)d_in[3];
    const float* bq = (const float*)d_in[4];
    const float* Wk = (const float*)d_in[5];
    const float* bk = (const float*)d_in[6];
    const float* Wv = (const float*)d_in[7];
    const float* bv = (const float*)d_in[8];
    const float* Wo = (const float*)d_in[9];
    const float* bo = (const float*)d_in[10];

    float* out      = (float*)d_out;                         // [B*L*D] = 4M
    float* attn_out = out + (size_t)NB * LSEQ * D_MODEL;     // [H*B*L*L] = 64M

    char*  ws  = (char*)d_ws;
    size_t off = 0;
    auto ALLOC = [&](size_t elems) -> _Float16* {
        _Float16* p = (_Float16*)(ws + off);
        off += ((elems * sizeof(_Float16)) + 255) & ~(size_t)255;
        return p;
    };
    const size_t NQKV = (size_t)NB * LSEQ * D_MODEL;  // 4M elems
    const size_t NW   = (size_t)NH * DH * D_MODEL;    // 1M elems
    _Float16* Q16  = ALLOC(NQKV);
    _Float16* K16  = ALLOC(NQKV);
    _Float16* V16  = ALLOC(NQKV);
    _Float16* WqT  = ALLOC(NW);           // [h][e][d]
    _Float16* WkT  = ALLOC(NW);
    _Float16* WvT  = ALLOC(NW);
    _Float16* WoT  = ALLOC(NW);           // [n][d]
    _Float16* q16  = ALLOC(NQKV);         // [b][h][l][e]
    _Float16* k16  = ALLOC(NQKV);         // [b][h][l][e]
    _Float16* vT16 = ALLOC(NQKV);         // [b][h][e][l]
    _Float16* O16  = ALLOC(NQKV);         // [b][l][h*64+e]
    (void)in_sizes; (void)n_in; (void)out_size; (void)ws_size;

    // 1. f32 -> f16 inputs
    int n4 = (int)(NQKV / 4);
    cvt_f32_to_f16<<<(n4 + 255) / 256, 256, 0, stream>>>(Q, Q16, n4);
    cvt_f32_to_f16<<<(n4 + 255) / 256, 256, 0, stream>>>(K, K16, n4);
    cvt_f32_to_f16<<<(n4 + 255) / 256, 256, 0, stream>>>(V, V16, n4);

    // 2. weight transposes: Wq/Wk/Wv [h][1024][64] -> [h][64][1024]; Wo -> WoT
    transpose_cvt<<<dim3(256, NH), 256, 0, stream>>>(Wq, WqT, D_MODEL, DH);
    transpose_cvt<<<dim3(256, NH), 256, 0, stream>>>(Wk, WkT, D_MODEL, DH);
    transpose_cvt<<<dim3(256, NH), 256, 0, stream>>>(Wv, WvT, D_MODEL, DH);
    transpose_cvt<<<dim3(4096, 1), 256, 0, stream>>>(Wo, WoT, D_MODEL, D_MODEL);

    // 3. projections: 3 matrices * 256 m-tiles * 16 head-blocks = 12288 waves
    proj_kernel<<<1536, 256, 0, stream>>>(Q16, K16, V16, WqT, WkT, WvT,
                                          bq, bk, bv, q16, k16, vT16);

    // 4. fused attention: one block per (b,h,16-query tile) = 4*16*64 = 4096
    attn_kernel<<<4096, 256, 0, stream>>>(q16, k16, vT16, attn_out, O16);

    // 5. output projection: 256 m-tiles * 16 n-blocks = 4096 waves
    outproj_kernel<<<512, 256, 0, stream>>>(O16, WoT, bo, out);
}
